// MagLaplacianEncoder_46145128628992
// MI455X (gfx1250) — matrix operand order repaired and checked by reading, not compiled, not run on hardware
//
#include <hip/hip_runtime.h>
#include <hip/hip_bf16.h>

typedef __attribute__((ext_vector_type(16))) _Float16 v16h;
typedef __attribute__((ext_vector_type(8)))  float    v8f;

#define NN 32768
#define KK 16
#define DE 32
#define HH 4
#define DA 256
#define WAVES 4
#define NODES_PER_WG 16   // WAVES * 4 nodes per workgroup

// Wave-local LDS ordering: DS ops from one wave complete in order (CDNA5 ISA
// §7.3), so a split-counter wait + compiler memory fence is all that is needed
// for per-wave staging buffers (no workgroup barrier).
__device__ __forceinline__ void lds_fence() {
    asm volatile("s_wait_dscnt 0" ::: "memory");
}

__launch_bounds__(128, 1)
__global__ void maglap_encoder_kernel(const float* __restrict__ eigval,
                                      const float* __restrict__ evre,
                                      const float* __restrict__ evim,
                                      const float* __restrict__ Welem,
                                      const float* __restrict__ belem,
                                      const float* __restrict__ Wq,
                                      const float* __restrict__ Wk,
                                      const float* __restrict__ Wv,
                                      const float* __restrict__ Wo,
                                      const float* __restrict__ bo,
                                      const float* __restrict__ Waggr,
                                      const float* __restrict__ baggr,
                                      float* __restrict__ out)
{
    // ---- shared (cross-wave) ----
    __shared__ float    sWelem[2][DE];
    __shared__ float    sbelem[DE];
    __shared__ float    sbo[DE];
    __shared__ float    sbaggr[DA];
    __shared__ float    spm[KK];                 // pmask as 0/1 float
    __shared__ _Float16 X[NODES_PER_WG][KK*DE];  // 16 x 512 f16 = 16 KB

    // ---- per-wave staging (wave-local; ordered with lds_fence) ----
    __shared__ float    trans32[WAVES][KK][DE];  // SignNet output (residual src)
    __shared__ _Float16 q16 [WAVES][KK][DE];
    __shared__ _Float16 k16 [WAVES][KK][DE];
    __shared__ _Float16 vv16[WAVES][KK][DE];
    __shared__ _Float16 wts [WAVES][KK][2*KK];   // softmax weights, K-padded to 32
    __shared__ _Float16 ah16[WAVES][KK][DE];     // per-head attention tile

    const int tid  = threadIdx.x;
    const int lane = tid & 31;
    const int wave = tid >> 5;
    const int g    = lane >> 4;      // half-wave group (0/1)
    const int lrow = lane & 15;

    // small constants into LDS + zero the K-pad of wts once
    if (tid < 64) sWelem[tid >> 5][tid & 31] = Welem[tid];
    if (tid < DE) { sbelem[tid] = belem[tid]; sbo[tid] = bo[tid]; }
    for (int i = tid; i < DA; i += blockDim.x) sbaggr[i] = baggr[i];
    if (tid < KK) spm[tid] = (eigval[tid] > 0.f || tid == 0) ? 1.f : 0.f;
    #pragma unroll
    for (int r = 0; r < 8; ++r) wts[wave][r + 8*g][16 + lrow] = (_Float16)0.f;
    __syncthreads();

    const int   node_base = blockIdx.x * NODES_PER_WG;
    const float rsqD = 0.17677669529663688f;   // 1/sqrt(32)

    for (int ni = 0; ni < 4; ++ni) {
        const int node_local = wave * 4 + ni;
        const int node = node_base + node_local;

        // ---- SignNet element MLP: relu(xW+b) + relu(-xW+b) -> trans32[16][32]
        {
            const int   t  = lrow;
            const float re = evre[(size_t)node * KK + t];
            const float im = evim[(size_t)node * KK + t];
            #pragma unroll
            for (int i = 0; i < 16; ++i) {
                const int   d   = g * 16 + i;
                const float lin = re * sWelem[0][d] + im * sWelem[1][d];
                const float b   = sbelem[d];
                trans32[wave][t][d] = fmaxf(lin + b, 0.f) + fmaxf(-lin + b, 0.f);
            }
        }
        lds_fence();

        // A-fragment of trans (16x32, K = d_in = 32) — reused for q/k/v, all heads
        v16h aT;
        #pragma unroll
        for (int i = 0; i < 8; ++i) {
            aT[i]     = (_Float16)trans32[wave][lrow][8*g + i];
            aT[i + 8] = (_Float16)trans32[wave][lrow][16 + 8*g + i];
        }

        // Wo-projected output accumulates across heads: out = sum_h attn_h @ Wo_h
        v8f cOut0 = {}, cOut1 = {};

        for (int h = 0; h < HH; ++h) {
            // ---- q/k/v head tiles: [16,32] each = 2 WMMAs each
            #pragma unroll
            for (int nh = 0; nh < 2; ++nh) {
                const int col = h * 32 + nh * 16;
                v16h bq, bk, bv;
                #pragma unroll
                for (int j = 0; j < 16; ++j) {
                    const int kr = (j < 8) ? 8*g + j : 8 + 8*g + j;
                    bq[j] = (_Float16)Wq[kr * 128 + col + lrow];  // B[K=d_in=kr][N=lrow]
                    bk[j] = (_Float16)Wk[kr * 128 + col + lrow];
                    bv[j] = (_Float16)Wv[kr * 128 + col + lrow];
                }
                v8f cq = {}, ck = {}, cv = {};
                cq = __builtin_amdgcn_wmma_f32_16x16x32_f16(false, aT, false, bq, (short)0, cq, false, false);
                ck = __builtin_amdgcn_wmma_f32_16x16x32_f16(false, aT, false, bk, (short)0, ck, false, false);
                cv = __builtin_amdgcn_wmma_f32_16x16x32_f16(false, aT, false, bv, (short)0, cv, false, false);
                #pragma unroll
                for (int r = 0; r < 8; ++r) {
                    const int m = r + 8 * g;
                    const int n = nh * 16 + lrow;
                    q16 [wave][m][n] = (_Float16)cq[r];
                    k16 [wave][m][n] = (_Float16)ck[r];
                    vv16[wave][m][n] = (_Float16)cv[r];
                }
            }
            lds_fence();

            // ---- logits = q @ k^T / sqrt(D): one WMMA (K = d = 32)
            v8f cl = {};
            {
                v16h aq, bkT;
                #pragma unroll
                for (int i = 0; i < 8; ++i) {
                    aq[i]     = q16[wave][lrow][8*g + i];
                    aq[i + 8] = q16[wave][lrow][16 + 8*g + i];
                }
                #pragma unroll
                for (int j = 0; j < 16; ++j)
                    bkT[j] = k16[wave][lrow][(j < 8) ? 8*g + j : 8 + 8*g + j];     // B[K=d][N=T=lrow] = k[T][d]
                cl = __builtin_amdgcn_wmma_f32_16x16x32_f16(false, aq, false, bkT, (short)0, cl, false, false);
            }

            // ---- softmax over T, fully in registers.
            // C layout: VGPR r holds row m=r+8g, column n=lrow, spread over a
            // 16-lane half-group -> row reductions are shfl_xor within width 16.
            {
                const float keep = spm[lrow];        // column (T) mask
                #pragma unroll
                for (int r = 0; r < 8; ++r) {
                    float v = (keep > 0.f) ? cl[r] * rsqD : -1e30f;
                    float mx = v;
                    mx = fmaxf(mx, __shfl_xor(mx, 1, 16));
                    mx = fmaxf(mx, __shfl_xor(mx, 2, 16));
                    mx = fmaxf(mx, __shfl_xor(mx, 4, 16));
                    mx = fmaxf(mx, __shfl_xor(mx, 8, 16));
                    const float e = __expf(v - mx);
                    float s = e;
                    s += __shfl_xor(s, 1, 16);
                    s += __shfl_xor(s, 2, 16);
                    s += __shfl_xor(s, 4, 16);
                    s += __shfl_xor(s, 8, 16);
                    wts[wave][r + 8*g][lrow] = (_Float16)(e / s);  // pad cols stay 0
                }
            }
            lds_fence();

            // ---- attn_h = w @ v : contraction T=16 zero-padded to 32, 2 WMMAs
            v16h aw;
            #pragma unroll
            for (int i = 0; i < 8; ++i) {
                aw[i]     = wts[wave][lrow][8*g + i];
                aw[i + 8] = wts[wave][lrow][16 + 8*g + i];
            }
            #pragma unroll
            for (int nh = 0; nh < 2; ++nh) {
                v16h bvv;
                #pragma unroll
                for (int j = 0; j < 16; ++j) {
                    const int kr = (j < 8) ? 8*g + j : 8 + 8*g + j;
                    _Float16 bval = (_Float16)0.f;             // rows T>=16 are padding
                    if (kr < 16) bval = vv16[wave][kr][nh * 16 + lrow];
                    bvv[j] = bval;                             // B[K=T=kr][N=d=lrow]
                }
                v8f ca = {};
                ca = __builtin_amdgcn_wmma_f32_16x16x32_f16(false, aw, false, bvv, (short)0, ca, false, false);
                #pragma unroll
                for (int r = 0; r < 8; ++r)
                    ah16[wave][r + 8*g][nh * 16 + lrow] = (_Float16)ca[r];
            }
            lds_fence();

            // ---- accumulate this head into out: attn_h[16,32] @ Wo_h[32,32]
            {
                v16h aA;
                #pragma unroll
                for (int i = 0; i < 8; ++i) {
                    aA[i]     = ah16[wave][lrow][8*g + i];
                    aA[i + 8] = ah16[wave][lrow][16 + 8*g + i];
                }
                v16h b0, b1;
                #pragma unroll
                for (int j = 0; j < 16; ++j) {
                    const int kr = (j < 8) ? 8*g + j : 8 + 8*g + j;
                    b0[j] = (_Float16)Wo[(h * 32 + kr) * 32 + lrow];       // N = 0..15
                    b1[j] = (_Float16)Wo[(h * 32 + kr) * 32 + 16 + lrow];  // N = 16..31
                }
                cOut0 = __builtin_amdgcn_wmma_f32_16x16x32_f16(false, aA, false, b0, (short)0, cOut0, false, false);
                cOut1 = __builtin_amdgcn_wmma_f32_16x16x32_f16(false, aA, false, b1, (short)0, cOut1, false, false);
            }
        } // heads

        // ---- residual + bias + row mask -> X (f16 input of aggregate GEMM)
        #pragma unroll
        for (int r = 0; r < 8; ++r) {
            const int m = r + 8 * g;
            const float r0 = trans32[wave][m][lrow]      + cOut0[r] + sbo[lrow];
            const float r1 = trans32[wave][m][16 + lrow] + cOut1[r] + sbo[16 + lrow];
            X[node_local][m * DE + lrow]      = (_Float16)(r0 * spm[m]);
            X[node_local][m * DE + 16 + lrow] = (_Float16)(r1 * spm[m]);
        }
    } // ni (nodes per wave)

    __syncthreads();   // all 16 node rows of X are final (cross-wave)

    // ---- aggregate MLP, batched over 16 nodes:
    // Y[16,256] = relu(X[16,512] @ W_aggr[512,256] + b)
    // K-chunks outer: each A-fragment is built ONCE, then feeds 4 independent
    // accumulating WMMAs (this wave's 4 column tiles) -> 4x less LDS A-traffic
    // and back-to-back independent WMMAs per K-step.
    {
        __builtin_prefetch(&Waggr[(size_t)lane * DA + wave * 16], 0, 1);
        v8f acc[4] = {v8f{}, v8f{}, v8f{}, v8f{}};
        for (int kc = 0; kc < 16; ++kc) {
            v16h a;
            #pragma unroll
            for (int i = 0; i < 8; ++i) {
                a[i]     = X[lrow][32*kc + 8*g + i];
                a[i + 8] = X[lrow][32*kc + 16 + 8*g + i];
            }
            const float* wrow = &Waggr[(size_t)(32*kc) * DA];
            #pragma unroll
            for (int t = 0; t < 4; ++t) {
                const int colb = (wave + 4*t) * 16;       // this wave's tiles
                v16h b;
                #pragma unroll
                for (int j = 0; j < 16; ++j)
                    b[j] = (_Float16)wrow[(size_t)((j < 8) ? 8*g + j : 8 + 8*g + j) * DA + colb + lrow];
                acc[t] = __builtin_amdgcn_wmma_f32_16x16x32_f16(false, a, false, b, (short)0, acc[t], false, false);
            }
        }
        #pragma unroll
        for (int t = 0; t < 4; ++t) {
            const int colb = (wave + 4*t) * 16;
            #pragma unroll
            for (int r = 0; r < 8; ++r) {
                const int m = r + 8 * g;                 // node row within workgroup
                const int n = colb + lrow;
                out[(size_t)(node_base + m) * DA + n] = fmaxf(acc[t][r] + sbaggr[n], 0.f);
            }
        }
    }
}

extern "C" void kernel_launch(void* const* d_in, const int* in_sizes, int n_in,
                              void* d_out, int out_size, void* d_ws, size_t ws_size,
                              hipStream_t stream) {
    (void)in_sizes; (void)n_in; (void)out_size; (void)d_ws; (void)ws_size;
    // dict order: senders(0), receivers(1) unused (use_gnn=False)
    const float* eigval = (const float*)d_in[2];
    const float* evre   = (const float*)d_in[3];
    const float* evim   = (const float*)d_in[4];
    const float* Welem  = (const float*)d_in[5];
    const float* belem  = (const float*)d_in[6];
    const float* Wq     = (const float*)d_in[7];
    const float* Wk     = (const float*)d_in[8];
    const float* Wv     = (const float*)d_in[9];
    const float* Wo     = (const float*)d_in[10];
    const float* bo     = (const float*)d_in[11];
    const float* Waggr  = (const float*)d_in[12];
    const float* baggr  = (const float*)d_in[13];
    float* out = (float*)d_out;

    dim3 grid(NN / NODES_PER_WG);   // 2048 workgroups of 16 nodes
    dim3 block(128);                // 4 wave32s
    maglap_encoder_kernel<<<grid, block, 0, stream>>>(
        eigval, evre, evim, Welem, belem, Wq, Wk, Wv, Wo, bo, Waggr, baggr, out);
}
